// BasicMoe_21500606284004
// MI455X (gfx1250) — matrix operand order repaired and checked by reading, not compiled, not run on hardware
//
#include <hip/hip_runtime.h>

#define N_TOK 4096
#define DIM   1024
#define NEXP  8
#define DFF   2048

typedef __bf16 bf16;
typedef __attribute__((ext_vector_type(16))) __bf16        v16bf;
typedef __attribute__((ext_vector_type(8)))  float         v8f;
typedef __attribute__((ext_vector_type(4)))  unsigned int  u32x4;
typedef __attribute__((ext_vector_type(8)))  int           i32x8;
typedef __attribute__((ext_vector_type(4)))  int           i32x4;
typedef __attribute__((ext_vector_type(4)))  float         f32x4;

#define TM 128
#define TN 128
#define TK 32
#define LSTR 40   // LDS row stride in halfs (80B = 20 DWORDs: 16 data + 4 pad)

// ---------------- Router: logits -> softmax -> top2 -> slot assignment ----
__global__ __launch_bounds__(256) void moe_router(
    const float* __restrict__ x, const float* __restrict__ gw,
    int* __restrict__ cnt, int* __restrict__ toke, int* __restrict__ toks,
    float* __restrict__ tokw, int* __restrict__ list)
{
    __shared__ float g[NEXP * DIM];
    int tid = threadIdx.x;
    for (int i = tid * 4; i < NEXP * DIM; i += 256 * 4)
        *(f32x4*)&g[i] = *(const f32x4*)&gw[i];
    __syncthreads();

    int n = blockIdx.x * 256 + tid;
    float acc[NEXP];
#pragma unroll
    for (int e = 0; e < NEXP; e++) acc[e] = 0.f;
    const float* xr = x + (size_t)n * DIM;
    for (int d = 0; d < DIM; d += 4) {
        f32x4 xv = *(const f32x4*)&xr[d];
#pragma unroll
        for (int e = 0; e < NEXP; e++) {
            const float* ge = &g[e * DIM + d];
            acc[e] += xv.x * ge[0] + xv.y * ge[1] + xv.z * ge[2] + xv.w * ge[3];
        }
    }
    // softmax (normalizer cancels in top-2 renorm, exp keeps monotone map)
    float m = acc[0];
#pragma unroll
    for (int e = 1; e < NEXP; e++) m = fmaxf(m, acc[e]);
    float p[NEXP];
#pragma unroll
    for (int e = 0; e < NEXP; e++) p[e] = __expf(acc[e] - m);
    int e0 = 0; float v0 = p[0];
#pragma unroll
    for (int e = 1; e < NEXP; e++) if (p[e] > v0) { v0 = p[e]; e0 = e; }
    int e1 = -1; float v1 = -1.f;
#pragma unroll
    for (int e = 0; e < NEXP; e++) if (e != e0 && p[e] > v1) { v1 = p[e]; e1 = e; }
    float inv = __builtin_amdgcn_rcpf(v0 + v1);
    float w0 = v0 * inv, w1 = v1 * inv;

    int s0 = atomicAdd(&cnt[e0], 1);
    int s1 = atomicAdd(&cnt[e1], 1);
    toke[2 * n + 0] = e0;  toke[2 * n + 1] = e1;
    toks[2 * n + 0] = s0;  toks[2 * n + 1] = s1;
    tokw[2 * n + 0] = w0;  tokw[2 * n + 1] = w1;
    list[e0 * N_TOK + s0] = n;
    list[e1 * N_TOK + s1] = n;
}

// ---------------- 8-entry exclusive prefix sum --------------------------
__global__ void moe_prefix(const int* __restrict__ cnt, int* __restrict__ base)
{
    if (threadIdx.x == 0) {
        int a = 0;
        for (int e = 0; e < NEXP; e++) { base[e] = a; a += cnt[e]; }
    }
}

// ---------------- FFN GEMM1: H = silu(gather(x) @ W1_e), bf16 out --------
__global__ __launch_bounds__(256) void moe_ffn1(
    const float* __restrict__ x, const float* __restrict__ w1,
    const int* __restrict__ cnt, const int* __restrict__ base,
    const int* __restrict__ list, bf16* __restrict__ H)
{
    __shared__ __align__(16) bf16 As[TM * LSTR];
    __shared__ __align__(16) bf16 Bs[TN * LSTR];
    __shared__ int toksh[TM];

    int e = blockIdx.z;
    int rows = cnt[e];
    int m0 = blockIdx.y * TM;
    if (m0 >= rows) return;
    int n0 = blockIdx.x * TN;
    int tid = threadIdx.x;
    int lane = tid & 31, wid = tid >> 5;
    int wm = (wid >> 2) * 64;
    int wn = (wid & 3) * 32;

    if (tid < TM) {
        int r = m0 + tid;
        int rc = r < rows ? r : rows - 1;
        toksh[tid] = list[e * N_TOK + rc];
    }
    __syncthreads();

    // per-thread staging sources
    int ar = tid >> 1;                 // A row 0..127
    int aseg = (tid & 1) * 16;         // halfs 0 or 16
    const float* asrc = x + (size_t)toksh[ar] * DIM + aseg;
    int bk = tid >> 3;                 // B k-row 0..31
    int bn = (tid & 7) * 16;           // B n offset 0..112
    const float* w1e = w1 + (size_t)e * DIM * DFF;

    v8f acc[4][2];
#pragma unroll
    for (int mi = 0; mi < 4; mi++)
#pragma unroll
        for (int ni = 0; ni < 2; ni++)
#pragma unroll
            for (int i = 0; i < 8; i++) acc[mi][ni][i] = 0.f;

    for (int kk = 0; kk < DIM; kk += TK) {
        __syncthreads();
        { // stage A: 128x32 f32 -> bf16 (gathered rows)
            f32x4 a0 = *(const f32x4*)(asrc + kk + 0);
            f32x4 a1 = *(const f32x4*)(asrc + kk + 4);
            f32x4 a2 = *(const f32x4*)(asrc + kk + 8);
            f32x4 a3 = *(const f32x4*)(asrc + kk + 12);
            union { bf16 h[16]; u32x4 q[2]; } pk;
            pk.h[0]=(bf16)a0.x; pk.h[1]=(bf16)a0.y; pk.h[2]=(bf16)a0.z; pk.h[3]=(bf16)a0.w;
            pk.h[4]=(bf16)a1.x; pk.h[5]=(bf16)a1.y; pk.h[6]=(bf16)a1.z; pk.h[7]=(bf16)a1.w;
            pk.h[8]=(bf16)a2.x; pk.h[9]=(bf16)a2.y; pk.h[10]=(bf16)a2.z; pk.h[11]=(bf16)a2.w;
            pk.h[12]=(bf16)a3.x; pk.h[13]=(bf16)a3.y; pk.h[14]=(bf16)a3.z; pk.h[15]=(bf16)a3.w;
            bf16* dst = &As[ar * LSTR + aseg];
            *(u32x4*)(dst + 0) = pk.q[0];
            *(u32x4*)(dst + 8) = pk.q[1];
        }
        { // stage B transposed: Bs[n][k] = w1[kk+k][n0+n], f32 -> bf16
            const float* src = w1e + (size_t)(kk + bk) * DFF + n0 + bn;
            __builtin_prefetch(src + (size_t)TK * DFF, 0, 0);
            f32x4 b0 = *(const f32x4*)(src + 0);
            f32x4 b1 = *(const f32x4*)(src + 4);
            f32x4 b2 = *(const f32x4*)(src + 8);
            f32x4 b3 = *(const f32x4*)(src + 12);
            float bv[16] = { b0.x,b0.y,b0.z,b0.w, b1.x,b1.y,b1.z,b1.w,
                             b2.x,b2.y,b2.z,b2.w, b3.x,b3.y,b3.z,b3.w };
#pragma unroll
            for (int i = 0; i < 16; i++) Bs[(bn + i) * LSTR + bk] = (bf16)bv[i];
        }
        __syncthreads();

        int half = lane >> 4;
        int l15 = lane & 15;
        v16bf af[4];
#pragma unroll
        for (int mi = 0; mi < 4; mi++) {
            const bf16* p = &As[(wm + mi * 16 + l15) * LSTR + half * 8];
            union { v16bf v; u32x4 q[2]; } u;
            u.q[0] = *(const u32x4*)(p + 0);    // K 0..7  (or 8..15)
            u.q[1] = *(const u32x4*)(p + 16);   // K 16..23 (or 24..31)
            af[mi] = u.v;
        }
        v16bf bfrag[2];
#pragma unroll
        for (int ni = 0; ni < 2; ni++) {
            const bf16* p = &Bs[(wn + ni * 16 + l15) * LSTR + half * 8];
            union { v16bf v; u32x4 q[2]; } u;
            u.q[0] = *(const u32x4*)(p + 0);    // K 0..7  (or 8..15)
            u.q[1] = *(const u32x4*)(p + 16);   // K 16..23 (or 24..31)
            bfrag[ni] = u.v;
        }
#pragma unroll
        for (int mi = 0; mi < 4; mi++)
#pragma unroll
            for (int ni = 0; ni < 2; ni++)
                acc[mi][ni] = __builtin_amdgcn_wmma_f32_16x16x32_bf16(
                    false, af[mi], false, bfrag[ni], (short)0, acc[mi][ni],
                    false, false);
    }

    // epilogue: SiLU (fast rcp), store bf16 H at compact offset base[e]+row.
    // One branch per fragment; fast path does 8 unguarded stores with
    // immediate r*DFF offsets off a single hoisted base pointer.
    int gbase = base[e];
    int half = lane >> 4;
    int l15 = lane & 15;
#pragma unroll
    for (int mi = 0; mi < 4; mi++) {
        int rbase = m0 + wm + mi * 16 + half * 8;
        int nv = rows - rbase;                 // valid rows among this 8
#pragma unroll
        for (int ni = 0; ni < 2; ni++) {
            int col = n0 + wn + ni * 16 + l15;
            bf16* hp = H + (size_t)(gbase + rbase) * DFF + col;
            if (nv >= 8) {
#pragma unroll
                for (int r = 0; r < 8; r++) {
                    float v = acc[mi][ni][r];
                    v = v * __builtin_amdgcn_rcpf(1.f + __expf(-v));
                    hp[(size_t)r * DFF] = (bf16)v;
                }
            } else {
#pragma unroll
                for (int r = 0; r < 8; r++) {
                    if (r < nv) {
                        float v = acc[mi][ni][r];
                        v = v * __builtin_amdgcn_rcpf(1.f + __expf(-v));
                        hp[(size_t)r * DFF] = (bf16)v;
                    }
                }
            }
        }
    }
}

// ---------------- FFN GEMM2: Y = H @ W2_e, f32 out -----------------------
// A tile staged by the Tensor Data Mover (contiguous compact H rows),
// with hardware zero-fill for ragged M tiles and LDS pad to LSTR stride.
__global__ __launch_bounds__(256) void moe_ffn2(
    const bf16* __restrict__ H, const float* __restrict__ w2,
    const int* __restrict__ cnt, const int* __restrict__ base,
    float* __restrict__ Y)
{
    __shared__ __align__(16) bf16 As[TM * LSTR];
    __shared__ __align__(16) bf16 Bs[TN * LSTR];

    int e = blockIdx.z;
    int rows = cnt[e];
    int m0 = blockIdx.y * TM;
    if (m0 >= rows) return;
    int n0 = blockIdx.x * TN;
    int tid = threadIdx.x;
    int lane = tid & 31, wid = tid >> 5;
    int wm = (wid >> 2) * 64;
    int wn = (wid & 3) * 32;
    int gbase = base[e];

    int bk = tid >> 3;
    int bn = (tid & 7) * 16;
    const float* w2e = w2 + (size_t)e * DFF * DIM;

    // TDM descriptor invariants (wave-uniform -> SGPRs)
    const bf16* atile = H + (size_t)(gbase + m0) * DFF;     // first row of tile
    unsigned lds_as = (unsigned)(uintptr_t)(void*)As;       // LDS byte offset
    int td1 = rows - m0;                                    // valid rows (>=1)

    v8f acc[4][2];
#pragma unroll
    for (int mi = 0; mi < 4; mi++)
#pragma unroll
        for (int ni = 0; ni < 2; ni++)
#pragma unroll
            for (int i = 0; i < 8; i++) acc[mi][ni][i] = 0.f;

    for (int kk = 0; kk < DFF; kk += TK) {
        __syncthreads();
        if (wid == 0) {
            // D# group0: count=1 | lds_addr | 57b global_addr | type=2
            uint64_t ga = (uint64_t)(uintptr_t)(const void*)(atile + kk);
            u32x4 g0;
            g0.x = 1u;
            g0.y = lds_as;
            g0.z = (unsigned)ga;
            g0.w = (unsigned)((ga >> 32) & 0x01FFFFFFull) | (2u << 30);
            // D# group1: data_size=2B, pad_enable, pad every 16 DW by 4 DW,
            // tensor 32 x td1 (zero-fill beyond), tile 32 x 128, stride DFF
            i32x8 g1;
            g1[0] = (int)((1u << 16) | (1u << 20) | (3u << 22) | (3u << 25));
            g1[1] = (int)(32u << 16);                              // tensor_dim0 lo16
            g1[2] = (int)(((unsigned)td1 & 0xFFFFu) << 16);        // dim0 hi=0 | dim1 lo16
            g1[3] = (int)(((unsigned)td1 >> 16) | (32u << 16));    // dim1 hi | tile_dim0=32
            g1[4] = 128;                                           // tile_dim1=128, tile_dim2=0
            g1[5] = DFF;                                           // tensor_dim0_stride lo32
            g1[6] = 0;
            g1[7] = 0;
            i32x4 z4 = { 0, 0, 0, 0 };
            i32x8 z8 = { 0, 0, 0, 0, 0, 0, 0, 0 };
            __builtin_amdgcn_tensor_load_to_lds(g0, g1, z4, z4, z8, 0);
        }
        { // stage B transposed: Bs[n][k] = w2[kk+k][n0+n], f32 -> bf16
            const float* src = w2e + (size_t)(kk + bk) * DIM + n0 + bn;
            __builtin_prefetch(src + (size_t)TK * DIM, 0, 0);
            f32x4 b0 = *(const f32x4*)(src + 0);
            f32x4 b1 = *(const f32x4*)(src + 4);
            f32x4 b2 = *(const f32x4*)(src + 8);
            f32x4 b3 = *(const f32x4*)(src + 12);
            float bv[16] = { b0.x,b0.y,b0.z,b0.w, b1.x,b1.y,b1.z,b1.w,
                             b2.x,b2.y,b2.z,b2.w, b3.x,b3.y,b3.z,b3.w };
#pragma unroll
            for (int i = 0; i < 16; i++) Bs[(bn + i) * LSTR + bk] = (bf16)bv[i];
        }
        if (wid == 0) __builtin_amdgcn_s_wait_tensorcnt(0);
        __syncthreads();

        int half = lane >> 4;
        int l15 = lane & 15;
        v16bf af[4];
#pragma unroll
        for (int mi = 0; mi < 4; mi++) {
            const bf16* p = &As[(wm + mi * 16 + l15) * LSTR + half * 8];
            union { v16bf v; u32x4 q[2]; } u;
            u.q[0] = *(const u32x4*)(p + 0);
            u.q[1] = *(const u32x4*)(p + 16);
            af[mi] = u.v;
        }
        v16bf bfrag[2];
#pragma unroll
        for (int ni = 0; ni < 2; ni++) {
            const bf16* p = &Bs[(wn + ni * 16 + l15) * LSTR + half * 8];
            union { v16bf v; u32x4 q[2]; } u;
            u.q[0] = *(const u32x4*)(p + 0);
            u.q[1] = *(const u32x4*)(p + 16);
            bfrag[ni] = u.v;
        }
#pragma unroll
        for (int mi = 0; mi < 4; mi++)
#pragma unroll
            for (int ni = 0; ni < 2; ni++)
                acc[mi][ni] = __builtin_amdgcn_wmma_f32_16x16x32_bf16(
                    false, af[mi], false, bfrag[ni], (short)0, acc[mi][ni],
                    false, false);
    }

    int half = lane >> 4;
    int l15 = lane & 15;
#pragma unroll
    for (int mi = 0; mi < 4; mi++) {
        int rbase = m0 + wm + mi * 16 + half * 8;
        int nv = rows - rbase;
#pragma unroll
        for (int ni = 0; ni < 2; ni++) {
            int col = n0 + wn + ni * 16 + l15;
            float* yp = Y + (size_t)(gbase + rbase) * DIM + col;
            if (nv >= 8) {
#pragma unroll
                for (int r = 0; r < 8; r++)
                    yp[(size_t)r * DIM] = acc[mi][ni][r];
            } else {
#pragma unroll
                for (int r = 0; r < 8; r++)
                    if (r < nv) yp[(size_t)r * DIM] = acc[mi][ni][r];
            }
        }
    }
}

// ---------------- Combine: out[n] = w0*Y[pos0] + w1*Y[pos1] --------------
__global__ __launch_bounds__(256) void moe_combine(
    const float* __restrict__ Y, const int* __restrict__ base,
    const int* __restrict__ toke, const int* __restrict__ toks,
    const float* __restrict__ tokw, float* __restrict__ out)
{
    int n = blockIdx.x;
    int e0 = toke[2 * n + 0], e1 = toke[2 * n + 1];
    int p0 = base[e0] + toks[2 * n + 0];
    int p1 = base[e1] + toks[2 * n + 1];
    float w0 = tokw[2 * n + 0], w1 = tokw[2 * n + 1];
    int c = threadIdx.x * 4;
    f32x4 y0 = *(const f32x4*)&Y[(size_t)p0 * DIM + c];
    f32x4 y1 = *(const f32x4*)&Y[(size_t)p1 * DIM + c];
    f32x4 o;
    o.x = w0 * y0.x + w1 * y1.x;
    o.y = w0 * y0.y + w1 * y1.y;
    o.z = w0 * y0.z + w1 * y1.z;
    o.w = w0 * y0.w + w1 * y1.w;
    *(f32x4*)&out[(size_t)n * DIM + c] = o;
}

extern "C" void kernel_launch(void* const* d_in, const int* in_sizes, int n_in,
                              void* d_out, int out_size, void* d_ws, size_t ws_size,
                              hipStream_t stream)
{
    const float* x  = (const float*)d_in[0];   // [N, D]
    const float* gw = (const float*)d_in[1];   // [E, D]
    const float* w1 = (const float*)d_in[2];   // [E, D, DFF]
    const float* w2 = (const float*)d_in[3];   // [E, DFF, D]
    float* out = (float*)d_out;                // [N, D]

    char* ws = (char*)d_ws;
    size_t off = 0;
    auto alloc = [&](size_t bytes) -> void* {
        void* p = ws + off;
        off = (off + bytes + 255) & ~(size_t)255;
        return p;
    };
    int*   cnt   = (int*)  alloc(NEXP * sizeof(int));
    int*   basep = (int*)  alloc(NEXP * sizeof(int));
    int*   toke  = (int*)  alloc((size_t)2 * N_TOK * sizeof(int));
    int*   toksl = (int*)  alloc((size_t)2 * N_TOK * sizeof(int));
    float* tokw  = (float*)alloc((size_t)2 * N_TOK * sizeof(float));
    int*   list  = (int*)  alloc((size_t)NEXP * N_TOK * sizeof(int));
    bf16*  Hbuf  = (bf16*) alloc((size_t)2 * N_TOK * DFF * sizeof(bf16));
    float* Ybuf  = (float*)alloc((size_t)2 * N_TOK * DIM * sizeof(float));
    (void)ws_size; (void)in_sizes; (void)n_in; (void)out_size;

    (void)hipMemsetAsync(cnt, 0, NEXP * sizeof(int), stream);
    moe_router<<<N_TOK / 256, 256, 0, stream>>>(x, gw, cnt, toke, toksl, tokw, list);
    moe_prefix<<<1, 32, 0, stream>>>(cnt, basep);
    dim3 g1(DFF / TN, N_TOK / TM, NEXP);
    moe_ffn1<<<g1, 256, 0, stream>>>(x, w1, cnt, basep, list, Hbuf);
    dim3 g2(DIM / TN, N_TOK / TM, NEXP);
    moe_ffn2<<<g2, 256, 0, stream>>>(Hbuf, w2, cnt, basep, Ybuf);
    moe_combine<<<N_TOK, 256, 0, stream>>>(Ybuf, basep, toke, toksl, tokw, out);
}
